// SnnModel_85907935854629
// MI455X (gfx1250) — matrix operand order repaired and checked by reading, not compiled, not run on hardware
//
#include <hip/hip_runtime.h>
#include <hip/hip_bf16.h>
#include <stdint.h>

// ---------------- CDNA5 vector types ----------------
typedef __attribute__((ext_vector_type(16))) _Float16 v16h;
typedef __attribute__((ext_vector_type(8)))  _Float16 v8h;
typedef __attribute__((ext_vector_type(8)))  float    v8f;
typedef __attribute__((ext_vector_type(2)))  float    v2f;

// ---------------- problem constants ----------------
#define TSTEPS 20
#define DECAY  0.875f
#define C1 64
#define D1 58
#define C2 128
#define D2 52
#define NFLAT 346112          // 128*52*52
#define NHID 512
#define NOUT 10
#define K2TOT 3136            // 64*49, conv2 reduction length
#define KHALF 1568            // 49 blocks of 32 per staging phase
#define SP1 3364              // 58*58 (conv1 plane)
#define SP2 2704              // 52*52 (conv2 plane)
#define SCOLS 32              // N for the big GEMM (20 used, padded to 2 WMMA tiles)
#define KCHUNK 2048           // 346112 = 169 * 2048 exactly
#define NKCHUNK 169

// ---------------- workspace layout (floats, then halves, then ktab u32) ----------------
#define OFF_MPIN  0
#define OFF_SIN   4096
#define OFF_MPC1  8192
#define OFF_MPC2  223488
#define OFF_Y     569600
#define OFF_S     585984
#define FLOAT_WORDS 11661568ull
#define HALF_F1SZ 215296
#define HALF_TOTAL (2*215296 + 128*3136)              // 832000 halves = 416000 words
#define KTAB_WORD_OFF (FLOAT_WORDS + HALF_TOTAL/2)
#define TOTAL_WS_WORDS (KTAB_WORD_OFF + K2TOT)

// ---------------- init: zero all state + d_out row 0 ----------------
__global__ void init_zero(float* wsf, float* out) {
    size_t i = (size_t)blockIdx.x * blockDim.x + threadIdx.x;
    size_t stride = (size_t)gridDim.x * blockDim.x;
    for (; i < TOTAL_WS_WORDS; i += stride) wsf[i] = 0.0f;
    if (blockIdx.x == 0 && threadIdx.x < (TSTEPS + 1) * NOUT)
        out[threadIdx.x] = 0.0f;
}

// ---------------- k -> (cin,ky,kx) lookup table: (cin*SP1 + ky*58) | (kx<<24) ----------------
__global__ void build_ktab(uint32_t* __restrict__ ktab) {
    int k = blockIdx.x * 256 + threadIdx.x;
    if (k < K2TOT) {
        int cin = k / 49;
        int rr  = k - cin * 49;
        int ky  = rr / 7;
        int kx  = rr - ky * 7;
        ktab[k] = (uint32_t)(cin * SP1 + ky * D1) | ((uint32_t)kx << 24);
    }
}

// ---------------- cast conv2 weights f32 -> f16 (layout already [128][3136]) ----------------
__global__ void w2_cast(const float* __restrict__ w, _Float16* __restrict__ wh) {
    int i = blockIdx.x * 256 + threadIdx.x;
    if (i < C2 * K2TOT) wh[i] = (_Float16)w[i];
}

// ---------------- input layer: integrate, fire, subtract ----------------
__global__ void input_step(const float* __restrict__ image,
                           float* __restrict__ mp_in, float* __restrict__ s_in) {
    int i = blockIdx.x * 256 + threadIdx.x;
    if (i < 4096) {
        float mp = mp_in[i] + image[i];
        float f = (mp >= 1.0f) ? 1.0f : 0.0f;
        mp_in[i] = mp - f;
        s_in[i] = f;
    }
}

// ---------------- conv1: 1->64 ch, 7x7 VALID, fire + decay; tiny, VALU ----------------
__global__ void conv1_step(const float* __restrict__ s_in, const float* __restrict__ w_c1,
                           float* __restrict__ mp_c1, _Float16* __restrict__ f1out) {
    int ox = threadIdx.x;          // 0..63
    int oy = blockIdx.x;           // 0..57
    int c  = blockIdx.y;           // 0..63
    if (ox >= D1) return;
    const float* w = w_c1 + c * 49;
    float sum = 0.0f;
#pragma unroll
    for (int ky = 0; ky < 7; ++ky)
#pragma unroll
        for (int kx = 0; kx < 7; ++kx)
            sum += w[ky * 7 + kx] * s_in[(oy + ky) * 64 + ox + kx];
    int idx = c * SP1 + oy * D1 + ox;
    float mp = mp_c1[idx] + sum;
    float f = (mp >= 1.0f) ? 1.0f : 0.0f;
    mp = (mp - f) * (f > 0.0f ? 1.0f : DECAY);
    mp_c1[idx] = mp;
    f1out[idx] = (_Float16)f;
}

// ---------------- conv2: implicit GEMM with V_WMMA_F32_16X16X32_F16 ----------------
// Grid: (4 ox-tiles, 52 oy). Block: 256 thr = 8 waves; wave w owns Cout 16w..16w+15.
// im2col B (shared by all waves) built cooperatively in LDS in fragment order,
// with all index decomposition pre-baked into the global ktab table (no divisions).
__global__ void conv2_step(const _Float16* __restrict__ f1prev,
                           const _Float16* __restrict__ w2h,
                           const uint32_t* __restrict__ ktab,
                           float* __restrict__ mp_c2,
                           float* __restrict__ Scol, int store_s) {
    // Bmat[kb][hi][n][j] : lane (hi*16+n) of k-block kb holds 16 contiguous halves j=0..15
    __shared__ _Float16 Bmat[49 * 512];   // 50,176 B
    int tid = threadIdx.x;
    int ox0 = blockIdx.x * 16;
    int oy  = blockIdx.y;

    int lane = tid & 31;
    int wv   = tid >> 5;
    int m0   = wv * 16;
    int n    = lane & 15;
    int hi   = lane >> 4;

    v8f acc = {};
    // A: 16-bit A 16x32 layout (ISA 7.12.2): lane group offset h = hi*8;
    // elems 0..7 -> K h..h+7, elems 8..15 -> K h+16..h+23.
    const _Float16* arow  = w2h + (size_t)(m0 + n) * K2TOT + hi * 8;
    const v16h* bfrag     = (const v16h*)&Bmat[(size_t)hi * 256 + n * 16];
    const _Float16* f1row = f1prev + oy * D1;   // ktab offset already has cin*SP1 + ky*58

    for (int phase = 0; phase < 2; ++phase) {
        int kpbase = phase * KHALF;
        // ---- cooperative im2col build, directly in fragment order ----
        for (int f = tid; f < 49 * 512; f += 256) {
            int kb  = f >> 9;           // k-block 0..48
            int r   = f & 511;
            int fhi = r >> 8;
            int fn  = (r >> 4) & 15;
            int j   = r & 15;
            uint32_t e = ktab[kpbase + kb * 32 + fhi * 8 + (j & 7) + ((j >> 3) << 4)];
            int kx = (int)(e >> 24);
            int x  = ox0 + kx + fn;             // may exceed plane for edge tiles
            _Float16 v = (_Float16)0.0f;
            if (x < D1) v = f1row[(e & 0xFFFFFFu) + x];
            Bmat[f] = v;
        }
        __syncthreads();
        // ---- hot loop: 49 WMMAs per wave per phase ----
        for (int kb = 0; kb < 49; ++kb) {
            int kk = kpbase + kb * 32;
            v8h alo = *(const v8h*)(arow + kk);
            v8h ahi = *(const v8h*)(arow + kk + 16);
            v16h a;
#pragma unroll
            for (int j = 0; j < 8; ++j) { a[j] = alo[j]; a[8 + j] = ahi[j]; }
            v16h b = bfrag[kb * 32];    // element stride: kb*512 halves
            acc = __builtin_amdgcn_wmma_f32_16x16x32_f16(
                false, a, false, b, (short)0, acc, false, false);
        }
        __syncthreads();
    }

    // C/D layout: lane n (0-15) -> N=n; hi lanes -> M+8; VGPR r -> M row.
    int ox = ox0 + n;
    if (ox < D2) {
        int mb = m0 + hi * 8;
#pragma unroll
        for (int r = 0; r < 8; ++r) {
            int idx = (mb + r) * SP2 + oy * D2 + ox;    // flat f2 index == S row
            float mp = mp_c2[idx] + acc[r];
            float fire = (mp >= 1.0f) ? 1.0f : 0.0f;
            mp = (mp - fire) * (fire > 0.0f ? 1.0f : DECAY);
            mp_c2[idx] = mp;
            if (store_s) Scol[(size_t)idx * SCOLS] = fire;
        }
    }
}

// ---------------- linear1 as ONE f32 GEMM: Y[512][32] = W @ S, WMMA 16x16x4 f32 ----------------
// Grid: (32 M-tiles, 169 K-chunks). Block: 1 wave. K-split partials via atomicAdd f32.
// HBM-bound on the single 709 MB pass of W; S (44 MB) is L2-resident.
__global__ void lin1_gemm(const float* __restrict__ W, const float* __restrict__ S,
                          float* __restrict__ Y) {
    int lane = threadIdx.x & 31;
    int m0 = blockIdx.x * 16;
    size_t kbase = (size_t)blockIdx.y * KCHUNK;

    int n  = lane & 15;
    int hi = lane >> 4;
    // f32 A 16x4 layout: lanes 0-15 carry K k,k+1; lanes 16-31 carry k+2,k+3.
    const float* wrow = W + (size_t)(m0 + n) * NFLAT;

    v8f acc0 = {};
    v8f acc1 = {};
    for (int kk = 0; kk < KCHUNK; kk += 4) {
        size_t k = kbase + kk + hi * 2;
        v2f a = *(const v2f*)(wrow + k);
        v2f b0, b1;
        b0[0] = S[k * SCOLS + n];
        b0[1] = S[(k + 1) * SCOLS + n];
        b1[0] = S[k * SCOLS + 16 + n];
        b1[1] = S[(k + 1) * SCOLS + 16 + n];
        acc0 = __builtin_amdgcn_wmma_f32_16x16x4_f32(
            false, a, false, b0, (short)0, acc0, false, false);
        acc1 = __builtin_amdgcn_wmma_f32_16x16x4_f32(
            false, a, false, b1, (short)0, acc1, false, false);
    }
    int mb = m0 + hi * 8;
#pragma unroll
    for (int r = 0; r < 8; ++r) {
        atomicAdd(&Y[(mb + r) * SCOLS + n],      acc0[r]);
        atomicAdd(&Y[(mb + r) * SCOLS + 16 + n], acc1[r]);
    }
}

// ---------------- sequential linear recurrences (20 steps, one block) ----------------
__global__ void heads_step(const float* __restrict__ Y, const float* __restrict__ w_l2,
                           float* __restrict__ out) {
    __shared__ float sfl1[NHID];
    int tid = threadIdx.x;
    float mp1 = 0.0f, mp2 = 0.0f;
    sfl1[tid] = 0.0f;
    __syncthreads();
    for (int t = 0; t < TSTEPS; ++t) {
        // linear2 reads fl1 from PREVIOUS step (sfl1 still holds t-1 value)
        if (tid < NOUT) {
            float acc = mp2 * DECAY;
            for (int j = 0; j < NHID; ++j) acc += w_l2[tid * NHID + j] * sfl1[j];
            float f = (acc >= 1.0f) ? 1.0f : 0.0f;
            out[(t + 1) * NOUT + tid] = f;
            mp2 = acc - f;
        }
        __syncthreads();
        // linear1: decay, add Y[:,t] (= W @ f2[t-1]), fire
        mp1 = mp1 * DECAY + Y[tid * SCOLS + t];
        float f = (mp1 >= 1.0f) ? 1.0f : 0.0f;
        mp1 -= f;
        sfl1[tid] = f;
        __syncthreads();
    }
}

// ---------------- launch ----------------
extern "C" void kernel_launch(void* const* d_in, const int* in_sizes, int n_in,
                              void* d_out, int out_size, void* d_ws, size_t ws_size,
                              hipStream_t stream) {
    (void)in_sizes; (void)n_in; (void)out_size; (void)ws_size;
    const float* image = (const float*)d_in[0];
    const float* w_c1  = (const float*)d_in[1];
    const float* w_c2  = (const float*)d_in[2];
    const float* w_l1  = (const float*)d_in[3];
    const float* w_l2  = (const float*)d_in[4];
    float* out = (float*)d_out;

    float* wsf   = (float*)d_ws;
    float* mp_in = wsf + OFF_MPIN;
    float* s_in  = wsf + OFF_SIN;
    float* mp_c1 = wsf + OFF_MPC1;
    float* mp_c2 = wsf + OFF_MPC2;
    float* Y     = wsf + OFF_Y;
    float* S     = wsf + OFF_S;
    _Float16* hbase = (_Float16*)(wsf + FLOAT_WORDS);
    _Float16* f1buf = hbase;                    // 2 x 215296
    _Float16* w2h   = hbase + 2 * HALF_F1SZ;    // 128*3136
    uint32_t* ktab  = (uint32_t*)(wsf + KTAB_WORD_OFF);

    init_zero<<<4096, 256, 0, stream>>>(wsf, out);
    build_ktab<<<(K2TOT + 255) / 256, 256, 0, stream>>>(ktab);
    w2_cast<<<(C2 * K2TOT + 255) / 256, 256, 0, stream>>>(w_c2, w2h);

    for (int t = 0; t < TSTEPS; ++t) {
        _Float16* f1_w = f1buf + (t & 1) * HALF_F1SZ;         // written this step
        _Float16* f1_r = f1buf + ((t & 1) ^ 1) * HALF_F1SZ;   // spikes from t-1
        input_step<<<16, 256, 0, stream>>>(image, mp_in, s_in);
        conv1_step<<<dim3(D1, C1), 64, 0, stream>>>(s_in, w_c1, mp_c1, f1_w);
        // S column t+1 holds f2[t]; linear1 at step t reads column t (= f2[t-1])
        conv2_step<<<dim3(4, D2), 256, 0, stream>>>(
            f1_r, w2h, ktab, mp_c2, S + (t + 1), (t < TSTEPS - 1) ? 1 : 0);
    }

    lin1_gemm<<<dim3(NHID / 16, NKCHUNK), 32, 0, stream>>>(w_l1, S, Y);
    heads_step<<<1, NHID, 0, stream>>>(Y, w_l2, out);
}
